// FeedbackRefinementNet_85495618995016
// MI455X (gfx1250) — matrix operand order repaired and checked by reading, not compiled, not run on hardware
//
#include <hip/hip_runtime.h>
#include <hip/hip_bf16.h>
#include <cstdint>

typedef __attribute__((ext_vector_type(16))) _Float16 v16h;
typedef __attribute__((ext_vector_type(8)))  float    v8f;

#define BN_EPS 1e-5f
#define NEG_INF (-3.4e38f)

// ======================================================================
// Generic GEMM with bias:  C[M,N] = A[M,K] @ W[K,N] + b[N]
// fp32 in/out, f16 WMMA compute (v_wmma_f32_16x16x32_f16), f32 accumulate.
// Block = 128 threads = 4 waves; each block owns a 64-row strip of A.
// Fragments are pre-swizzled into LDS in per-lane order, contiguous k = 16*(lane/16) + i for
// both A and B (not the ISA table order): each lane pulls its whole v16h
// fragment as 32 contiguous bytes (2x ds_load_b128). K loop fully unrolled
// via the KC template; A fragments are register-resident across all N
// tiles; two N tiles run with independent accumulators (no WMMA->WMMA RAW
// hazard). Stores take a uniform full-tile fast path (M is a multiple of
// 64 for every GEMM in this network) -> plain global_store_b32, no
// per-lane exec-mask churn.
// ======================================================================
#define GEMM_ROWS 64
#define GEMM_MAXN 128

template<int KC>   // number of K chunks of 32 (Kp = KC*32)
__global__ __launch_bounds__(128)
void gemm_f16wmma(const float* __restrict__ A, const float* __restrict__ W,
                  const float* __restrict__ bias, float* __restrict__ C,
                  long M, int K, int N) {
  constexpr int Kp = KC * 32;
  __shared__ _Float16 As[GEMM_ROWS * Kp];      // swizzled [r][c][hk][16]
  __shared__ _Float16 Ws[Kp * GEMM_MAXN];      // swizzled [c][t][lane][16]
  const int Np = (N + 15) & ~15;
  const int NT = Np >> 4;                      // number of 16-wide N tiles
  const int tid  = threadIdx.x;
  const int lane = tid & 31;
  const int wave = tid >> 5;
  const long rowBase = (long)blockIdx.x * GEMM_ROWS;

  // Prefetch next A strip into L2 (global_prefetch_b8 path)
  if (rowBase + GEMM_ROWS < M)
    __builtin_prefetch(&A[(rowBase + GEMM_ROWS) * (long)K], 0, 0);

  // Stage + swizzle W:
  //   dst ((c*NT + t)*32 + l)*16 + i  <-  W[c*32 + (l>>4)*16 + i][t*16 + (l&15)]
  for (int d = tid; d < KC * NT * 32 * 16; d += 128) {
    int i = d & 15;
    int l = (d >> 4) & 31;
    int rest = d >> 9;                 // c*NT + t
    int t = rest % NT, c = rest / NT;
    int k = c * 32 + ((l >> 4) << 4) + i;
    int n = (t << 4) + (l & 15);
    Ws[d] = (_Float16)((k < K && n < N) ? W[(long)k * N + n] : 0.0f);
  }
  // Stage + swizzle A:
  //   dst ((r*KC + c)*2 + hk)*16 + h  <-  A[row][c*32 + hk*16 + h]
  for (int d = tid; d < GEMM_ROWS * Kp; d += 128) {
    int h  = d & 15;
    int hk = (d >> 4) & 1;
    int rest = d >> 5;                 // r*KC + c
    int c = rest % KC, r = rest / KC;
    int k = c * 32 + hk * 16 + h;
    long row = rowBase + r;
    As[d] = (_Float16)((row < M && k < K) ? A[row * (long)K + k] : 0.0f);
  }
  __syncthreads();

  const int r16    = lane & 15;
  const int hk     = lane >> 4;
  const int rowL   = wave * 16 + r16;
  const int rowOff = (lane < 16) ? 0 : 8;
  const bool fullTile = (rowBase + GEMM_ROWS <= M);   // uniform per block

  // Register-resident A fragments for this wave's 16-row strip.
  v16h afrag[KC];
#pragma unroll
  for (int c = 0; c < KC; ++c)
    afrag[c] = *(const v16h*)&As[(((rowL * KC + c) * 2) + hk) * 16];

  auto storeTile = [&](const v8f& acc, int t) {
    const int col = (t << 4) + r16;
    if (col < N) {
      const float bv = bias[col];
      float* cp = C + (rowBase + wave * 16 + rowOff) * (long)N + col;
      if (fullTile) {
#pragma unroll
        for (int v = 0; v < 8; ++v) cp[(long)v * N] = acc[v] + bv;
      } else {
#pragma unroll
        for (int v = 0; v < 8; ++v) {
          long row = rowBase + wave * 16 + rowOff + v;
          if (row < M) C[row * (long)N + col] = acc[v] + bv;
        }
      }
    }
  };

  int nt = 0;
  for (; nt + 2 <= NT; nt += 2) {      // two tiles -> independent WMMA chains
    v8f acc0 = {}, acc1 = {};
#pragma unroll
    for (int c = 0; c < KC; ++c) {
      v16h b0 = *(const v16h*)&Ws[(((c * NT + nt)     * 32) + lane) * 16];
      v16h b1 = *(const v16h*)&Ws[(((c * NT + nt + 1) * 32) + lane) * 16];
      acc0 = __builtin_amdgcn_wmma_f32_16x16x32_f16(false, afrag[c], false, b0,
                                                    (short)0, acc0, false, false);
      acc1 = __builtin_amdgcn_wmma_f32_16x16x32_f16(false, afrag[c], false, b1,
                                                    (short)0, acc1, false, false);
    }
    storeTile(acc0, nt);
    storeTile(acc1, nt + 1);
  }
  if (nt < NT) {
    v8f acc = {};
#pragma unroll
    for (int c = 0; c < KC; ++c) {
      v16h b = *(const v16h*)&Ws[(((c * NT + nt) * 32) + lane) * 16];
      acc = __builtin_amdgcn_wmma_f32_16x16x32_f16(false, afrag[c], false, b,
                                                   (short)0, acc, false, false);
    }
    storeTile(acc, nt);
  }
}

// ======================================================================
// BatchNorm (training-style, stats over all rows) + leaky-relu, in place.
// ======================================================================
__global__ __launch_bounds__(256)
void bn_stats_kernel(const float* __restrict__ X, long M, int C,
                     float* __restrict__ mean, float* __restrict__ rstd) {
  const int c = blockIdx.x;            // one block per channel
  float s = 0.f, s2 = 0.f;
  for (long m = threadIdx.x; m < M; m += 256) {
    float x = X[m * C + c];
    s += x; s2 += x * x;
  }
  __shared__ float ls[256], lq[256];
  ls[threadIdx.x] = s; lq[threadIdx.x] = s2;
  __syncthreads();
  for (int o = 128; o > 0; o >>= 1) {
    if (threadIdx.x < o) {
      ls[threadIdx.x] += ls[threadIdx.x + o];
      lq[threadIdx.x] += lq[threadIdx.x + o];
    }
    __syncthreads();
  }
  if (threadIdx.x == 0) {
    float mu  = ls[0] / (float)M;
    float var = lq[0] / (float)M - mu * mu;
    mean[c] = mu;
    rstd[c] = rsqrtf(var + BN_EPS);
  }
}

__global__ void bn_act_kernel(float* __restrict__ X, long total, int C,
                              const float* __restrict__ mean, const float* __restrict__ rstd,
                              const float* __restrict__ g, const float* __restrict__ be,
                              float slope) {
  long i = (long)blockIdx.x * blockDim.x + threadIdx.x;
  if (i >= total) return;
  int c = (int)(i % C);
  float y = (X[i] - mean[c]) * rstd[c] * g[c] + be[c];
  X[i] = (y > 0.f) ? y : y * slope;   // slope=0 => relu, 0.2 => leaky
}

// ======================================================================
// Farthest point sampling over concat(A[b], B[b]); entirely in LDS.
// One block per batch. N <= 4096 => 64 KB LDS (fits 320 KB WGP LDS).
// ======================================================================
#define FPS_MAXN 4096
__global__ __launch_bounds__(256)
void fps_kernel(const float* __restrict__ A, int nA,
                const float* __restrict__ Bp, int nB,
                float* __restrict__ out, int m) {
  const int b = blockIdx.x;
  const int N = nA + nB;
  __shared__ float sx[FPS_MAXN], sy[FPS_MAXN], sz[FPS_MAXN], sd[FPS_MAXN];
  __shared__ float rv[256];
  __shared__ int   ri[256];
  __shared__ float scx, scy, scz;
  const float* pa = A  + (long)b * nA * 3;
  const float* pb = Bp + (long)b * nB * 3;
  for (int i = threadIdx.x; i < N; i += 256) {
    const float* p = (i < nA) ? (pa + (long)i * 3) : (pb + (long)(i - nA) * 3);
    sx[i] = p[0]; sy[i] = p[1]; sz[i] = p[2];
  }
  __syncthreads();
  const float x0 = sx[0], y0 = sy[0], z0 = sz[0];
  for (int i = threadIdx.x; i < N; i += 256) {
    float dx = sx[i]-x0, dy = sy[i]-y0, dz = sz[i]-z0;
    sd[i] = dx*dx + dy*dy + dz*dz;
  }
  if (threadIdx.x == 0) {
    float* o = out + (long)b * m * 3;
    o[0] = x0; o[1] = y0; o[2] = z0;
  }
  __syncthreads();
  for (int t = 1; t < m; ++t) {
    float bv = NEG_INF; int bi = 0;
    for (int i = threadIdx.x; i < N; i += 256) {
      float d = sd[i];
      if (d > bv) { bv = d; bi = i; }
    }
    rv[threadIdx.x] = bv; ri[threadIdx.x] = bi;
    __syncthreads();
    for (int o = 128; o > 0; o >>= 1) {
      if (threadIdx.x < o && rv[threadIdx.x + o] > rv[threadIdx.x]) {
        rv[threadIdx.x] = rv[threadIdx.x + o];
        ri[threadIdx.x] = ri[threadIdx.x + o];
      }
      __syncthreads();
    }
    if (threadIdx.x == 0) {
      int s = ri[0];
      scx = sx[s]; scy = sy[s]; scz = sz[s];
      float* o = out + ((long)b * m + t) * 3;
      o[0] = scx; o[1] = scy; o[2] = scz;
    }
    __syncthreads();
    const float px = scx, py = scy, pz = scz;
    for (int i = threadIdx.x; i < N; i += 256) {
      float dx = sx[i]-px, dy = sy[i]-py, dz = sz[i]-pz;
      float d = dx*dx + dy*dy + dz*dz;
      if (d < sd[i]) sd[i] = d;
    }
    __syncthreads();
  }
}

// ======================================================================
// kNN: one thread per query, reference points tiled through LDS,
// top-K kept as a sorted insertion list in registers.
// ======================================================================
template<int K, int D>
__global__ __launch_bounds__(128)
void knn_kernel(const float* __restrict__ Q, int nq,
                const float* __restrict__ R, int nr,
                int* __restrict__ idx) {
  const int b = blockIdx.y;
  const int q = blockIdx.x * 128 + threadIdx.x;
  const float* Qb = Q + (long)b * nq * D;
  const float* Rb = R + (long)b * nr * D;
  float qv[D];
#pragma unroll
  for (int d = 0; d < D; ++d) qv[d] = (q < nq) ? Qb[(long)q * D + d] : 0.f;
  float best[K]; int bidx[K];
#pragma unroll
  for (int i = 0; i < K; ++i) { best[i] = 3.4e38f; bidx[i] = 0; }
  __shared__ float tile[128 * D];
  for (int base = 0; base < nr; base += 128) {
    const int cnt = (128 < nr - base) ? 128 : (nr - base);
    __syncthreads();
    for (int i = threadIdx.x; i < cnt * D; i += 128) tile[i] = Rb[(long)base * D + i];
    __syncthreads();
    for (int j = 0; j < cnt; ++j) {
      float d2 = 0.f;
#pragma unroll
      for (int d = 0; d < D; ++d) { float df = qv[d] - tile[j * D + d]; d2 += df * df; }
      if (d2 < best[K - 1]) {
        int p = K - 1;
        while (p > 0 && best[p - 1] > d2) {
          best[p] = best[p - 1]; bidx[p] = bidx[p - 1]; --p;
        }
        best[p] = d2; bidx[p] = base + j;
      }
    }
  }
  if (q < nq) {
#pragma unroll
    for (int i = 0; i < K; ++i) idx[((long)b * nq + q) * K + i] = bidx[i];
  }
}

// ======================================================================
// EdgeConv feature build: out[(b,n,j)] = concat(x_i, x_j - x_i)   (width 2D)
// ======================================================================
__global__ void edge_feat_kernel(const float* __restrict__ X, const int* __restrict__ idx,
                                 float* __restrict__ out, int B, int N, int K, int D) {
  long i = (long)blockIdx.x * blockDim.x + threadIdx.x;
  long total = (long)B * N * K * D;
  if (i >= total) return;
  int d = (int)(i % D); long t = i / D;
  int j = (int)(t % K); long t2 = t / K;
  int n = (int)(t2 % N); int b = (int)(t2 / N);
  int jn = idx[((long)b * N + n) * K + j];
  float xi = X[((long)b * N + n)  * D + d];
  float xj = X[((long)b * N + jn) * D + d];
  long row = ((long)b * N + n) * K + j;
  out[row * 2 * D + d]     = xi;
  out[row * 2 * D + D + d] = xj - xi;
}

__global__ void max_pool_k_kernel(const float* __restrict__ H, float* __restrict__ out,
                                  long M, int K, int C) {
  long i = (long)blockIdx.x * blockDim.x + threadIdx.x;
  if (i >= M * C) return;
  long m = i / C; int c = (int)(i % C);
  float v = NEG_INF;
  for (int j = 0; j < K; ++j) v = fmaxf(v, H[(m * K + j) * C + c]);
  out[i] = v;
}

// feat(B*N,64) -> concat(feat, broadcast(rowmax))  (B*N,128)
__global__ void cat_fmax_kernel(const float* __restrict__ F, float* __restrict__ out,
                                long M, int C) {
  long i = (long)blockIdx.x * blockDim.x + threadIdx.x;
  if (i >= M * 2 * C) return;
  long m = i / (2 * C); int c = (int)(i % (2 * C));
  if (c < C) out[i] = F[m * C + c];
  else {
    float v = NEG_INF;
    for (int j = 0; j < C; ++j) v = fmaxf(v, F[m * C + j]);
    out[i] = v;
  }
}

__global__ void concat_rows_kernel(const float* __restrict__ A, int nA,
                                   const float* __restrict__ Bp, int nB,
                                   float* __restrict__ out, int B, int D) {
  long i = (long)blockIdx.x * blockDim.x + threadIdx.x;
  long total = (long)B * (nA + nB) * D;
  if (i >= total) return;
  int d = (int)(i % D); long t = i / D;
  int n = (int)(t % (nA + nB)); int b = (int)(t / (nA + nB));
  out[i] = (n < nA) ? A[((long)b * nA + n) * D + d]
                    : Bp[((long)b * nB + (n - nA)) * D + d];
}

// ---- cross transformer helpers (k fixed at 2) ----
__global__ void ct_pos_in_kernel(const float* __restrict__ px, const float* __restrict__ pf,
                                 const int* __restrict__ idx, float* __restrict__ out,
                                 int B, int M, int Nf) {
  long i = (long)blockIdx.x * blockDim.x + threadIdx.x;
  long total = (long)B * M * 2 * 3;
  if (i >= total) return;
  int d = (int)(i % 3); long t = i / 3;
  int j = (int)(t & 1); long t2 = t >> 1;
  int m = (int)(t2 % M); int b = (int)(t2 / M);
  int jn = idx[((long)b * M + m) * 2 + j];
  out[i] = px[((long)b * M + m) * 3 + d] - pf[((long)b * Nf + jn) * 3 + d];
}

__global__ void ct_attn_in_kernel(const float* __restrict__ fx, const float* __restrict__ ff,
                                  const int* __restrict__ idx, const float* __restrict__ delta,
                                  float* __restrict__ out, int B, int M, int Nf) {
  long i = (long)blockIdx.x * blockDim.x + threadIdx.x;
  long total = (long)B * M * 2 * 32;
  if (i >= total) return;
  int c = (int)(i % 32); long t = i / 32;
  int j = (int)(t & 1); long t2 = t >> 1;
  int m = (int)(t2 % M); int b = (int)(t2 / M);
  int jn = idx[((long)b * M + m) * 2 + j];
  out[i] = fx[((long)b * M + m) * 32 + c] - ff[((long)b * Nf + jn) * 32 + c] + delta[i];
}

// softmax over the 2 neighbors, weighted sum of (x_j + delta)
__global__ void ct_out_kernel(const float* __restrict__ attn, const float* __restrict__ delta,
                              const float* __restrict__ ff, const int* __restrict__ idx,
                              float* __restrict__ out, int B, int M, int Nf) {
  long i = (long)blockIdx.x * blockDim.x + threadIdx.x;
  long total = (long)B * M * 32;
  if (i >= total) return;
  int c = (int)(i % 32); long t = i / 32;
  int m = (int)(t % M); int b = (int)(t / M);
  long base = ((long)b * M + m) * 2;
  float a0 = attn[(base + 0) * 32 + c], a1 = attn[(base + 1) * 32 + c];
  int  j0 = idx[base + 0],              j1 = idx[base + 1];
  float v0 = ff[((long)b * Nf + j0) * 32 + c] + delta[(base + 0) * 32 + c];
  float v1 = ff[((long)b * Nf + j1) * 32 + c] + delta[(base + 1) * 32 + c];
  float mx = fmaxf(a0, a1);
  float e0 = __expf(a0 - mx), e1 = __expf(a1 - mx);
  out[i] = (e0 * v0 + e1 * v1) / (e0 + e1);
}

// out[b, n*up+r, :] = pcd[b, n, :] + delta[b, n*up+r, :]
__global__ void upsample_add_kernel(const float* __restrict__ pcd, const float* __restrict__ delta,
                                    float* __restrict__ out, int B, int N, int up) {
  long i = (long)blockIdx.x * blockDim.x + threadIdx.x;
  long total = (long)B * N * up * 3;
  if (i >= total) return;
  int d = (int)(i % 3); long t = i / 3;
  int n2 = (int)(t % (N * up)); int b = (int)(t / (N * up));
  out[i] = pcd[((long)b * N + n2 / up) * 3 + d] + delta[i];
}

// ======================================================================
// Host-side orchestration
// ======================================================================
struct P2 { const float *W1, *W2, *b1, *b2, *be1, *g1; };
struct P3 { const float *W1, *W2, *W3, *b1, *b2, *b3, *be1, *be2, *g1, *g2; };
struct BlockP { P2 ct_attn, ct_pos, gcn, mlp, ns; P3 md; };

struct Arena { char* p; size_t off; };
static float* af(Arena& a, size_t n) {
  float* r = (float*)(a.p + a.off);
  a.off += ((n * sizeof(float) + 255) & ~(size_t)255);
  return r;
}
static int* ai(Arena& a, size_t n) {
  int* r = (int*)(a.p + a.off);
  a.off += ((n * sizeof(int) + 255) & ~(size_t)255);
  return r;
}
static inline int cdiv(long a, int b) { return (int)((a + b - 1) / b); }

// JAX pytree flatten: dicts in sorted-key order.
// mlp2: W1, W2, b1, b2, be1, g1 ; mlp3: W1, W2, W3, b1, b2, b3, be1, be2, g1, g2
// block: ct_attn, ct_pos, gcn, md, mlp, ns
static P2 readP2(void* const* d, int& i) {
  P2 p;
  p.W1 = (const float*)d[i++]; p.W2 = (const float*)d[i++];
  p.b1 = (const float*)d[i++]; p.b2 = (const float*)d[i++];
  p.be1 = (const float*)d[i++]; p.g1 = (const float*)d[i++];
  return p;
}
static P3 readP3(void* const* d, int& i) {
  P3 p;
  p.W1 = (const float*)d[i++]; p.W2 = (const float*)d[i++]; p.W3 = (const float*)d[i++];
  p.b1 = (const float*)d[i++]; p.b2 = (const float*)d[i++]; p.b3 = (const float*)d[i++];
  p.be1 = (const float*)d[i++]; p.be2 = (const float*)d[i++];
  p.g1 = (const float*)d[i++]; p.g2 = (const float*)d[i++];
  return p;
}
static BlockP readBlock(void* const* d, int& i) {
  BlockP b;
  b.ct_attn = readP2(d, i);
  b.ct_pos  = readP2(d, i);
  b.gcn     = readP2(d, i);
  b.md      = readP3(d, i);
  b.mlp     = readP2(d, i);
  b.ns      = readP2(d, i);
  return b;
}

static void launch_gemm(const float* A, const float* W, const float* b, float* C,
                        long M, int K, int N, hipStream_t s) {
  const int grid = cdiv(M, GEMM_ROWS);
  const int kc = (K + 31) / 32;
  if (kc <= 1)      gemm_f16wmma<1><<<grid, 128, 0, s>>>(A, W, b, C, M, K, N);
  else if (kc == 2) gemm_f16wmma<2><<<grid, 128, 0, s>>>(A, W, b, C, M, K, N);
  else              gemm_f16wmma<4><<<grid, 128, 0, s>>>(A, W, b, C, M, K, N);
}
static void launch_bn_act(float* X, long M, int C, const float* g, const float* be,
                          float slope, float* stat, hipStream_t s) {
  bn_stats_kernel<<<C, 256, 0, s>>>(X, M, C, stat, stat + 128);
  long total = M * C;
  bn_act_kernel<<<cdiv(total, 256), 256, 0, s>>>(X, total, C, stat, stat + 128, g, be, slope);
}
static void run_mlp2(const float* X, long M, int c0, int c1, int c2, const P2& p,
                     float slope, float* H, float* Y, float* stat, hipStream_t s) {
  launch_gemm(X, p.W1, p.b1, H, M, c0, c1, s);
  launch_bn_act(H, M, c1, p.g1, p.be1, slope, stat, s);
  launch_gemm(H, p.W2, p.b2, Y, M, c1, c2, s);
}

template<int K, int D>
static void launch_knn(const float* Q, int nq, const float* R, int nr, int* idx,
                       int B, hipStream_t s) {
  dim3 grid((nq + 127) / 128, B);
  knn_kernel<K, D><<<grid, 128, 0, s>>>(Q, nq, R, nr, idx);
}

static void edge_conv(const float* X, int B, int N, int D, int k, int c1, int c2,
                      const P2& p, float* out, Arena& a, hipStream_t s) {
  Arena save = a;
  int* idx = ai(a, (size_t)B * N * k);
  if (D == 3)  launch_knn<16, 3>(X, N, X, N, idx, B, s);
  else         launch_knn<8, 32>(X, N, X, N, idx, B, s);
  long rows = (long)B * N * k;
  float* E = af(a, (size_t)rows * 2 * D);
  long tg = rows * D;
  edge_feat_kernel<<<cdiv(tg, 256), 256, 0, s>>>(X, idx, E, B, N, k, D);
  float* H    = af(a, (size_t)rows * c1);
  float* Hout = af(a, (size_t)rows * c2);
  float* stat = af(a, 256);
  run_mlp2(E, rows, 2 * D, c1, c2, p, 0.2f, H, Hout, stat, s);
  long t2 = (long)B * N * c2;
  max_pool_k_kernel<<<cdiv(t2, 256), 256, 0, s>>>(Hout, out, (long)B * N, k, c2);
  a = save;
}

static void cross_tf(const float* px, const float* fx, int N,
                     const float* py, const float* fy, int Nn,
                     const P2& pp, const P2& pa, int B, float* out,
                     Arena& a, hipStream_t s) {
  Arena save = a;
  int Nf = N + Nn;
  float* pf = af(a, (size_t)B * Nf * 3);
  float* ff = af(a, (size_t)B * Nf * 32);
  long tp = (long)B * Nf * 3;
  concat_rows_kernel<<<cdiv(tp, 256), 256, 0, s>>>(px, N, py, Nn, pf, B, 3);
  long tf = (long)B * Nf * 32;
  concat_rows_kernel<<<cdiv(tf, 256), 256, 0, s>>>(fx, N, fy, Nn, ff, B, 32);
  int* idx = ai(a, (size_t)B * N * 2);
  launch_knn<2, 3>(px, N, pf, Nf, idx, B, s);
  long rows = (long)B * N * 2;
  float* Pin = af(a, (size_t)rows * 3);
  ct_pos_in_kernel<<<cdiv(rows * 3, 256), 256, 0, s>>>(px, pf, idx, Pin, B, N, Nf);
  float* H     = af(a, (size_t)rows * 32);
  float* Delta = af(a, (size_t)rows * 32);
  float* stat  = af(a, 256);
  run_mlp2(Pin, rows, 3, 32, 32, pp, 0.0f, H, Delta, stat, s);
  float* Ain = af(a, (size_t)rows * 32);
  ct_attn_in_kernel<<<cdiv(rows * 32, 256), 256, 0, s>>>(fx, ff, idx, Delta, Ain, B, N, Nf);
  float* Attn = af(a, (size_t)rows * 32);
  run_mlp2(Ain, rows, 32, 32, 32, pa, 0.0f, H, Attn, stat, s);
  long to = (long)B * N * 32;
  ct_out_kernel<<<cdiv(to, 256), 256, 0, s>>>(Attn, Delta, ff, idx, out, B, N, Nf);
  a = save;
}

static void fbac(const float* pcd, int B, int N, const BlockP& blk, int up,
                 const float* pcd_next, int Nn, const float* feat_next,
                 float* out_pcd, float* out_feat, Arena& a, hipStream_t s) {
  Arena save = a;
  float* feat64 = af(a, (size_t)B * N * 64);
  edge_conv(pcd, B, N, 3, 16, 32, 64, blk.gcn, feat64, a, s);
  float* feat128 = af(a, (size_t)B * N * 128);
  long t128 = (long)B * N * 128;
  cat_fmax_kernel<<<cdiv(t128, 256), 256, 0, s>>>(feat64, feat128, (long)B * N, 64);
  float* feat32 = af(a, (size_t)B * N * 32);
  float* H      = af(a, (size_t)B * N * 128);
  float* stat   = af(a, 256);
  run_mlp2(feat128, (long)B * N, 128, 128, 32, blk.mlp, 0.0f, H, feat32, stat, s);
  const float* pn = pcd_next  ? pcd_next  : pcd;
  const float* fn = pcd_next  ? feat_next : feat32;
  int Nn2         = pcd_next  ? Nn        : N;
  float* ctout = af(a, (size_t)B * N * 32);
  cross_tf(pcd, feat32, N, pn, fn, Nn2, blk.ct_pos, blk.ct_attn, B, ctout, a, s);
  // ns EdgeConv: 32-d feature kNN (k=8), mlp2(64 -> 16*up -> 32*up), max over k.
  edge_conv(ctout, B, N, 32, 8, 16 * up, 32 * up, blk.ns, out_feat, a, s);
  // reshape (B,N,32*up) -> (B,N*up,32) is a pure reinterpretation (row-major).
  long rows = (long)B * N * up;
  float* H1 = af(a, (size_t)rows * 64);
  float* H2 = af(a, (size_t)rows * 64);
  float* dl = af(a, (size_t)rows * 3);
  launch_gemm(out_feat, blk.md.W1, blk.md.b1, H1, rows, 32, 64, s);
  launch_bn_act(H1, rows, 64, blk.md.g1, blk.md.be1, 0.0f, stat, s);
  launch_gemm(H1, blk.md.W2, blk.md.b2, H2, rows, 64, 64, s);
  launch_bn_act(H2, rows, 64, blk.md.g2, blk.md.be2, 0.0f, stat, s);
  launch_gemm(H2, blk.md.W3, blk.md.b3, dl, rows, 64, 3, s);
  upsample_add_kernel<<<cdiv(rows * 3, 256), 256, 0, s>>>(pcd, dl, out_pcd, B, N, up);
  a = save;
}

extern "C" void kernel_launch(void* const* d_in, const int* in_sizes, int n_in,
                              void* d_out, int out_size, void* d_ws, size_t ws_size,
                              hipStream_t stream) {
  (void)in_sizes; (void)n_in; (void)out_size; (void)ws_size;
  const float* pcd     = (const float*)d_in[0];
  const float* partial = (const float*)d_in[1];
  int cur = 2;
  BlockP b0 = readBlock(d_in, cur);
  BlockP b1 = readBlock(d_in, cur);

  const int B = 8, NIN = 2048, NPS = 1024;
  Arena a{ (char*)d_ws, 0 };
  float* pcd00  = af(a, (size_t)B * 2048 * 3);   // pcd_state[0][0]
  float* feat00 = af(a, (size_t)B * 2048 * 32);  // feat_state[0][0]
  float* pcd01  = af(a, (size_t)B * 4096 * 3);   // pcd_state[0][1]
  float* feat01 = af(a, (size_t)B * 4096 * 32);  // feat_state[0][1]
  float* pcdT   = af(a, (size_t)B * 2048 * 3);   // fps output temp
  float* featT  = af(a, (size_t)B * 4096 * 32);  // scratch feat for cycle 1

  float* outA = (float*)d_out;                          // (8,2048,3)
  float* outB = (float*)d_out + (size_t)B * 2048 * 3;   // (8,4096,3)

  // ---- cycle 0, block 0 ----
  fps_kernel<<<B, 256, 0, stream>>>(pcd, NIN, partial, NIN, pcdT, NPS);
  fbac(pcdT, B, 1024, b0, 2, nullptr, 0, nullptr, pcd00, feat00, a, stream);
  // ---- cycle 0, block 1 ----
  fps_kernel<<<B, 256, 0, stream>>>(pcd00, 2048, partial, NIN, pcdT, 2048);
  fbac(pcdT, B, 2048, b1, 2, nullptr, 0, nullptr, pcd01, feat01, a, stream);
  // ---- cycle 1, block 0 (feedback from cycle 0) ----
  fps_kernel<<<B, 256, 0, stream>>>(pcd00, 2048, partial, NIN, pcdT, NPS);
  fbac(pcdT, B, 1024, b0, 2, pcd00, 2048, feat00, outA, featT, a, stream);
  // ---- cycle 1, block 1 ----
  fps_kernel<<<B, 256, 0, stream>>>(outA, 2048, partial, NIN, pcdT, 2048);
  fbac(pcdT, B, 2048, b1, 2, pcd01, 4096, feat01, outB, featT, a, stream);
}
